// TransActor_2748779069600
// MI455X (gfx1250) — matrix operand order repaired and checked by reading, not compiled, not run on hardware
//
#include <hip/hip_runtime.h>
#include <hip/hip_bf16.h>

#define N_NODES 20000
#define N_EDGES 160000
#define F_INDIM 64
#define HID     256
#define HEADS   4
#define QKV     (HEADS * HID)   // 1024
#define OUT_DIM 2
#define BN_EPS  1e-5f

typedef __attribute__((ext_vector_type(16))) __bf16        v16bf;
typedef __attribute__((ext_vector_type(8)))  float         v8f;
typedef __attribute__((ext_vector_type(8)))  unsigned int  v8u;
typedef __attribute__((ext_vector_type(4)))  unsigned int  u32x4;
typedef __attribute__((ext_vector_type(8)))  int           i32x8;
typedef __attribute__((ext_vector_type(4)))  int           i32x4;

#if __has_builtin(__builtin_amdgcn_tensor_load_to_lds)
#define HAVE_TDM 1
#else
#define HAVE_TDM 0
#endif

// ---------------------------------------------------------------- helpers
__device__ __forceinline__ unsigned short f2bf(float f) {
  unsigned u = __float_as_uint(f);
  unsigned r = u + 0x7FFFu + ((u >> 16) & 1u);   // round-to-nearest-even
  return (unsigned short)(r >> 16);
}
// monotonic float<->uint encoding for atomicMax over signed floats
__device__ __forceinline__ unsigned fenc(float f) {
  unsigned u = __float_as_uint(f);
  return (u & 0x80000000u) ? ~u : (u | 0x80000000u);
}
__device__ __forceinline__ float fdec(unsigned u) {
  unsigned b = (u & 0x80000000u) ? (u & 0x7FFFFFFFu) : ~u;
  return __uint_as_float(b);
}

#if HAVE_TDM
// Issue a 2-D TDM tile load: tile (tile_d1 rows x tile_d0 elems) of a row-major
// tensor (stride0 elems between rows), 2-byte elements, into LDS at lds_off.
// Rows beyond tensor_d1 are OOB -> zero-filled by the TDM.
__device__ __forceinline__ void tdm_load_2d(unsigned lds_off, const void* gptr,
                                            unsigned tile_d0, unsigned tile_d1,
                                            unsigned tensor_d0, unsigned tensor_d1,
                                            unsigned stride0) {
  unsigned long long ga = (unsigned long long)gptr;
  u32x4 g0;
  g0[0] = 1u;                                      // count=1, user mode, no gather
  g0[1] = lds_off;                                 // lds_addr (bytes)
  g0[2] = (unsigned)(ga & 0xFFFFFFFFull);          // global_addr[31:0]
  g0[3] = (unsigned)((ga >> 32) & 0x01FFFFFFull)   // global_addr[56:32]
          | (2u << 30);                            // type = 2 (image)
  i32x8 g1;
  g1[0] = (int)(1u << 16);                                         // data_size=2B
  g1[1] = (int)((tensor_d0 & 0xFFFFu) << 16);                      // td0[15:0]
  g1[2] = (int)(((tensor_d0 >> 16) & 0xFFFFu) |
                ((tensor_d1 & 0xFFFFu) << 16));                    // td0[31:16] | td1[15:0]
  g1[3] = (int)(((tensor_d1 >> 16) & 0xFFFFu) |
                ((tile_d0 & 0xFFFFu) << 16));                      // td1[31:16] | tile_d0
  g1[4] = (int)(tile_d1 & 0xFFFFu);                                // tile_d1, tile_d2=0
  g1[5] = (int)stride0;                                            // dim0_stride[31:0]
  g1[6] = 0;                                                       // stride hi / dim1_stride
  g1[7] = 0;
  i32x4 z4 = { 0, 0, 0, 0 };
  i32x8 z8 = { 0, 0, 0, 0, 0, 0, 0, 0 };
  // amdgpu-toolchain (clang-23) 6-arg form: (g0, g1, g2, g3, g4, cpol)
  __builtin_amdgcn_tensor_load_to_lds(g0, g1, z4, z4, z8, 0);
}
#endif

// ---------------------------------------------------------------- fills
__global__ void fill_f32(float* p, float v, int n) {
  int i = blockIdx.x * blockDim.x + threadIdx.x;
  if (i < n) p[i] = v;
}
__global__ void fill_u32(unsigned* p, unsigned v, int n) {
  int i = blockIdx.x * blockDim.x + threadIdx.x;
  if (i < n) p[i] = v;
}

// ---------------------------------------------------------------- converts
__global__ void cvt_f32_bf16(const float* __restrict__ s, unsigned short* __restrict__ d, int n) {
  int i = blockIdx.x * blockDim.x + threadIdx.x;
  if (i < n) d[i] = f2bf(s[i]);
}
// W: K x Nc row-major  ->  Wt: Nc x K row-major (bf16)
__global__ void cvt_transpose_bf16(const float* __restrict__ W, unsigned short* __restrict__ Wt,
                                   int K, int Nc) {
  int i = blockIdx.x * blockDim.x + threadIdx.x;
  if (i < K * Nc) {
    int k = i / Nc, n = i % Nc;
    Wt[(size_t)n * K + k] = f2bf(W[i]);
  }
}

// ---------------------------------------------------------------- WMMA GEMM
// C[M x Nc] = A[M x K](bf16) * Bt[Nc x K](bf16)^T + bias, fp32 accumulate.
// Block tile 128x64, 8 waves (4M x 2N), each wave: 32x32 = 2x2 WMMA tiles.
#define BM 128
#define BN 64
#define BK 32
__global__ __launch_bounds__(256)
void gemm_bf16_wmma(const unsigned short* __restrict__ A,
                    const unsigned short* __restrict__ Bt,
                    const float* __restrict__ bias,
                    float* __restrict__ C, int M, int Nc, int K) {
  __shared__ unsigned short As[2][BM * BK];
  __shared__ unsigned short Bs[2][BN * BK];
  const int tid  = threadIdx.x;
  const int m0   = blockIdx.x * BM;
  const int n0   = blockIdx.y * BN;
  const int wave = tid >> 5;
  const int lane = tid & 31;
  const int half = lane >> 4;      // 0 | 1
  const int idx  = lane & 15;      // 0..15
  const int wm   = wave >> 1;      // 0..3  (32-row M sub-tile)
  const int wn   = wave & 1;       // 0..1  (32-col N sub-tile)

  v8f acc00 = {}, acc01 = {}, acc10 = {}, acc11 = {};
  const int nsteps = K / BK;

#if HAVE_TDM
  const bool issuer = (tid < 32);                        // wave 0 drives the TDM
  const unsigned rows_left = (unsigned)(M - m0);         // OOB rows -> zero fill
  if (issuer) {
    tdm_load_2d((unsigned)(unsigned long long)&As[0][0], A + (size_t)m0 * K,
                BK, BM, (unsigned)K, rows_left, (unsigned)K);
    tdm_load_2d((unsigned)(unsigned long long)&Bs[0][0], Bt + (size_t)n0 * K,
                BK, BN, (unsigned)K, (unsigned)(Nc - n0), (unsigned)K);
  }
#endif

  for (int i = 0; i < nsteps; ++i) {
    const int cur = i & 1;
#if HAVE_TDM
    if (issuer) {
      if (i + 1 < nsteps) {                              // prefetch next K-slice
        const int nxt = cur ^ 1;
        tdm_load_2d((unsigned)(unsigned long long)&As[nxt][0],
                    A + (size_t)m0 * K + (size_t)(i + 1) * BK,
                    BK, BM, (unsigned)K, rows_left, (unsigned)K);
        tdm_load_2d((unsigned)(unsigned long long)&Bs[nxt][0],
                    Bt + (size_t)n0 * K + (size_t)(i + 1) * BK,
                    BK, BN, (unsigned)K, (unsigned)(Nc - n0), (unsigned)K);
        __builtin_amdgcn_s_wait_tensorcnt(2);            // current slice resident
      } else {
        __builtin_amdgcn_s_wait_tensorcnt(0);
      }
    }
    __syncthreads();                                     // publish LDS to all waves
#else
    {   // per-thread staging fallback
      __syncthreads();
      const int lrow = tid >> 2;        // 0..63
      const int lcol = (tid & 3) * 8;   // 8 bf16 = 16 bytes
      int k0 = i * BK;
      for (int rr = 0; rr < BM; rr += 64) {
        uint4 av = make_uint4(0, 0, 0, 0);
        int gr = m0 + rr + lrow;
        if (gr < M) av = *(const uint4*)(A + (size_t)gr * K + k0 + lcol);
        *(uint4*)(&As[cur][(rr + lrow) * BK + lcol]) = av;
      }
      uint4 bv = make_uint4(0, 0, 0, 0);
      int gn = n0 + lrow;
      if (gn < Nc) bv = *(const uint4*)(Bt + (size_t)gn * K + k0 + lcol);
      *(uint4*)(&Bs[cur][lrow * BK + lcol]) = bv;
      __syncthreads();
    }
#endif

    // A fragments (16x32 bf16), per ISA layout:
    //   lane(half, m=idx): VGPR0-3 K=half*8..+7 ; VGPR4-7 K=16+half*8..+7
    v16bf afrag[2];
#pragma unroll
    for (int ms = 0; ms < 2; ++ms) {
      const unsigned short* arow = &As[cur][(wm * 32 + ms * 16 + idx) * BK];
      uint4 alo = *(const uint4*)(arow + half * 8);
      uint4 ahi = *(const uint4*)(arow + 16 + half * 8);
      v8u raw = { alo.x, alo.y, alo.z, alo.w, ahi.x, ahi.y, ahi.z, ahi.w };
      afrag[ms] = __builtin_bit_cast(v16bf, raw);
    }
    // B fragments (32x16): lane(half, n=idx): VGPR0-3 K=half*8..+7 ; VGPR4-7 K=16+half*8..+7
    v16bf bfrag[2];
#pragma unroll
    for (int ns = 0; ns < 2; ++ns) {
      const unsigned short* brow = &Bs[cur][(wn * 32 + ns * 16 + idx) * BK + half * 8];
      uint4 blo = *(const uint4*)(brow);
      uint4 bhi = *(const uint4*)(brow + 16);
      v8u raw = { blo.x, blo.y, blo.z, blo.w, bhi.x, bhi.y, bhi.z, bhi.w };
      bfrag[ns] = __builtin_bit_cast(v16bf, raw);
    }

    acc00 = __builtin_amdgcn_wmma_f32_16x16x32_bf16(false, afrag[0], false, bfrag[0],
                                                    (short)0, acc00, false, false);
    acc01 = __builtin_amdgcn_wmma_f32_16x16x32_bf16(false, afrag[0], false, bfrag[1],
                                                    (short)0, acc01, false, false);
    acc10 = __builtin_amdgcn_wmma_f32_16x16x32_bf16(false, afrag[1], false, bfrag[0],
                                                    (short)0, acc10, false, false);
    acc11 = __builtin_amdgcn_wmma_f32_16x16x32_bf16(false, afrag[1], false, bfrag[1],
                                                    (short)0, acc11, false, false);
    __syncthreads();                                     // all reads done before rewrite
  }

  // C/D layout: VGPR r, lanes0-15 -> M=r, lanes16-31 -> M=8+r, N=idx
  const int col0 = n0 + wn * 32 + idx;
  const int col1 = col0 + 16;
  const float bb0 = bias ? bias[col0] : 0.f;
  const float bb1 = bias ? bias[col1] : 0.f;
#pragma unroll
  for (int r = 0; r < 8; ++r) {
    int row0 = m0 + wm * 32 + half * 8 + r;
    int row1 = row0 + 16;
    if (row0 < M) {
      C[(size_t)row0 * Nc + col0] = acc00[r] + bb0;
      C[(size_t)row0 * Nc + col1] = acc01[r] + bb1;
    }
    if (row1 < M) {
      C[(size_t)row1 * Nc + col0] = acc10[r] + bb0;
      C[(size_t)row1 * Nc + col1] = acc11[r] + bb1;
    }
  }
}

// ---------------------------------------------------------------- attention
__global__ __launch_bounds__(256)
void edge_alpha(const float* __restrict__ q, const float* __restrict__ k,
                const int* __restrict__ src, const int* __restrict__ dst,
                float* __restrict__ alpha, unsigned* __restrict__ menc) {
  int i = blockIdx.x * blockDim.x + threadIdx.x;   // e*HEADS + h
  if (i >= N_EDGES * HEADS) return;
  int e = i >> 2, h = i & 3;
  int d = dst[e], s = src[e];
  const float4* qd = (const float4*)(q + (size_t)d * QKV + h * HID);
  const float4* ks = (const float4*)(k + (size_t)s * QKV + h * HID);
  float acc = 0.f;
#pragma unroll 8
  for (int j = 0; j < HID / 4; ++j) {
    float4 a = qd[j], b = ks[j];
    acc += a.x * b.x + a.y * b.y + a.z * b.z + a.w * b.w;
  }
  float al = acc * 0.0625f;   // / sqrt(256)
  alpha[i] = al;
  atomicMax(&menc[(size_t)d * HEADS + h], fenc(al));
}

__global__ void seg_max_fix(const unsigned* __restrict__ menc, float* __restrict__ mdec, int n) {
  int i = blockIdx.x * blockDim.x + threadIdx.x;
  if (i < n) {
    float f = fdec(menc[i]);
    mdec[i] = (f < -3.0e38f) ? 0.f : f;   // where(isfinite(m), m, 0)
  }
}

__global__ void edge_exp(float* __restrict__ alpha, const float* __restrict__ mdec,
                         const int* __restrict__ dst, float* __restrict__ denom) {
  int i = blockIdx.x * blockDim.x + threadIdx.x;
  if (i >= N_EDGES * HEADS) return;
  int e = i >> 2, h = i & 3;
  int d = dst[e];
  float ex = __expf(alpha[i] - mdec[(size_t)d * HEADS + h]);
  alpha[i] = ex;
  atomicAdd(&denom[(size_t)d * HEADS + h], ex);
}

// one wave per (edge, head): 32 lanes x 8 channels, coalesced float atomics
__global__ __launch_bounds__(256)
void edge_aggregate(const float* __restrict__ ex, const float* __restrict__ denom,
                    const float* __restrict__ v, const int* __restrict__ src,
                    const int* __restrict__ dst, float* __restrict__ agg) {
  int wid = (blockIdx.x * blockDim.x + threadIdx.x) >> 5;
  if (wid >= N_EDGES * HEADS) return;
  int lane = threadIdx.x & 31;
  int e = wid >> 2, h = wid & 3;
  int d = dst[e], s = src[e];
  float coeff = ex[wid] / (denom[(size_t)d * HEADS + h] + 1e-16f);
  const float* vp = v + (size_t)s * QKV + h * HID;
  float* ap = agg + (size_t)d * QKV + h * HID;
#pragma unroll
  for (int j = 0; j < 8; ++j) {
    int c = lane + j * 32;
    atomicAdd(&ap[c], coeff * vp[c]);
  }
}

__global__ void combine_heads(const float* __restrict__ agg, const float* __restrict__ skip,
                              float* __restrict__ xo, int n) {
  int i = blockIdx.x * blockDim.x + threadIdx.x;
  if (i >= n * HID) return;
  int node = i / HID, c = i % HID;
  const float* a = agg + (size_t)node * QKV;
  float s = a[c] + a[HID + c] + a[2 * HID + c] + a[3 * HID + c];
  xo[i] = s * 0.25f + skip[i];
}

// ---------------------------------------------------------------- batchnorm + head
__global__ __launch_bounds__(256)
void bn_reduce(const float* __restrict__ x, float* __restrict__ sums, int n) {
  int c = threadIdx.x;                                // channel
  int rows = (n + gridDim.x - 1) / gridDim.x;
  int r0 = blockIdx.x * rows;
  int r1 = r0 + rows; if (r1 > n) r1 = n;
  float s = 0.f, s2 = 0.f;
  for (int r = r0; r < r1; ++r) {
    float v = x[(size_t)r * HID + c];
    s += v; s2 += v * v;
  }
  atomicAdd(&sums[c], s);
  atomicAdd(&sums[HID + c], s2);
}

__global__ void bn_finalize(const float* __restrict__ sums, const float* __restrict__ gamma,
                            const float* __restrict__ beta, float* __restrict__ inv,
                            float* __restrict__ betap, float n) {
  int c = threadIdx.x;
  float mu  = sums[c] / n;
  float var = sums[HID + c] / n - mu * mu;
  float iv  = gamma[c] * rsqrtf(var + BN_EPS);
  inv[c]   = iv;
  betap[c] = beta[c] - mu * iv;
}

__global__ __launch_bounds__(256)
void head_kernel(const float* __restrict__ x, const float* __restrict__ inv,
                 const float* __restrict__ betap, const float* __restrict__ lw,
                 const float* __restrict__ lb, float* __restrict__ out, int n) {
  int node = blockIdx.x * blockDim.x + threadIdx.x;
  if (node >= n) return;
  const float* xr = x + (size_t)node * HID;
  float l0 = lb[0], l1 = lb[1];
  float* emb = out + (size_t)n * OUT_DIM;   // rsu_embedding after action_prob
  for (int c = 0; c < HID; ++c) {
    float y = xr[c] * inv[c] + betap[c];
    l0 += y * lw[c * OUT_DIM + 0];
    l1 += y * lw[c * OUT_DIM + 1];
    if (node == 0) emb[c] = y;
  }
  l0 = fmaxf(l0, 0.f); l1 = fmaxf(l1, 0.f);
  float mx = fmaxf(l0, l1);
  float e0 = __expf(l0 - mx), e1 = __expf(l1 - mx);
  float dsum = e0 + e1;
  out[node * 2 + 0] = e0 / dsum;
  out[node * 2 + 1] = e1 / dsum;
}

// ---------------------------------------------------------------- launch
static inline int cdiv(long long a, long long b) { return (int)((a + b - 1) / b); }

extern "C" void kernel_launch(void* const* d_in, const int* in_sizes, int n_in,
                              void* d_out, int out_size, void* d_ws, size_t ws_size,
                              hipStream_t stream) {
  (void)in_sizes; (void)n_in; (void)out_size; (void)ws_size;
  const float* xin = (const float*)d_in[0];
  const int*   ei  = (const int*)d_in[1];
  const int* src = ei;
  const int* dst = ei + N_EDGES;
  const float* wq1 = (const float*)d_in[2],  *bq1 = (const float*)d_in[3];
  const float* wk1 = (const float*)d_in[4],  *bk1 = (const float*)d_in[5];
  const float* wv1 = (const float*)d_in[6],  *bv1 = (const float*)d_in[7];
  const float* ws1 = (const float*)d_in[8],  *bs1 = (const float*)d_in[9];
  const float* wq2 = (const float*)d_in[10], *bq2 = (const float*)d_in[11];
  const float* wk2 = (const float*)d_in[12], *bk2 = (const float*)d_in[13];
  const float* wv2 = (const float*)d_in[14], *bv2 = (const float*)d_in[15];
  const float* ws2 = (const float*)d_in[16], *bs2 = (const float*)d_in[17];
  const float* gam = (const float*)d_in[18], *bet = (const float*)d_in[19];
  const float* lw  = (const float*)d_in[20], *lb  = (const float*)d_in[21];
  float* out = (float*)d_out;

  // ---- workspace carve-up (all offsets 256B-aligned)
  char* w = (char*)d_ws;
  float* q    = (float*)w;  w += (size_t)N_NODES * QKV * 4;
  float* kbuf = (float*)w;  w += (size_t)N_NODES * QKV * 4;
  float* v    = (float*)w;  w += (size_t)N_NODES * QKV * 4;
  float* agg  = (float*)w;  w += (size_t)N_NODES * QKV * 4;
  float* skip = (float*)w;  w += (size_t)N_NODES * HID * 4;
  float* xcur = (float*)w;  w += (size_t)N_NODES * HID * 4;
  unsigned short* xbf   = (unsigned short*)w; w += (size_t)N_NODES * HID * 2;
  unsigned short* wtbuf = (unsigned short*)w; w += 2u * 1024u * 1024u;
  float*    alpha = (float*)w;    w += (size_t)N_EDGES * HEADS * 4;
  unsigned* menc  = (unsigned*)w; w += (size_t)N_NODES * HEADS * 4;
  float*    mdec  = (float*)w;    w += (size_t)N_NODES * HEADS * 4;
  float*    denom = (float*)w;    w += (size_t)N_NODES * HEADS * 4;
  float*    bnsum = (float*)w;    w += 512 * 4;
  float*    bninv = (float*)w;    w += 256 * 4;
  float*    bnbet = (float*)w;    w += 256 * 4;

  auto run_layer = [&](const float* xf, int K,
                       const float* wq, const float* bq, const float* wk, const float* bk,
                       const float* wv, const float* bv, const float* wsk, const float* bsk,
                       float* xout) {
    const int nel = N_NODES * K;
    cvt_f32_bf16<<<cdiv(nel, 256), 256, 0, stream>>>(xf, xbf, nel);

    unsigned short* wqT = wtbuf;
    unsigned short* wkT = wqT + (size_t)QKV * K;
    unsigned short* wvT = wkT + (size_t)QKV * K;
    unsigned short* wsT = wvT + (size_t)QKV * K;
    cvt_transpose_bf16<<<cdiv((long long)K * QKV, 256), 256, 0, stream>>>(wq, wqT, K, QKV);
    cvt_transpose_bf16<<<cdiv((long long)K * QKV, 256), 256, 0, stream>>>(wk, wkT, K, QKV);
    cvt_transpose_bf16<<<cdiv((long long)K * QKV, 256), 256, 0, stream>>>(wv, wvT, K, QKV);
    cvt_transpose_bf16<<<cdiv((long long)K * HID, 256), 256, 0, stream>>>(wsk, wsT, K, HID);

    dim3 gq(cdiv(N_NODES, BM), QKV / BN);
    gemm_bf16_wmma<<<gq, 256, 0, stream>>>(xbf, wqT, bq, q,    N_NODES, QKV, K);
    gemm_bf16_wmma<<<gq, 256, 0, stream>>>(xbf, wkT, bk, kbuf, N_NODES, QKV, K);
    gemm_bf16_wmma<<<gq, 256, 0, stream>>>(xbf, wvT, bv, v,    N_NODES, QKV, K);
    dim3 gs(cdiv(N_NODES, BM), HID / BN);
    gemm_bf16_wmma<<<gs, 256, 0, stream>>>(xbf, wsT, bsk, skip, N_NODES, HID, K);

    // per-call re-init (graph-replay safe)
    fill_u32<<<cdiv(N_NODES * HEADS, 256), 256, 0, stream>>>(menc, 0x007FFFFFu /* enc(-inf) */,
                                                             N_NODES * HEADS);
    fill_f32<<<cdiv(N_NODES * HEADS, 256), 256, 0, stream>>>(denom, 0.f, N_NODES * HEADS);
    fill_f32<<<cdiv((long long)N_NODES * QKV, 256), 256, 0, stream>>>(agg, 0.f, N_NODES * QKV);

    edge_alpha<<<cdiv(N_EDGES * HEADS, 256), 256, 0, stream>>>(q, kbuf, src, dst, alpha, menc);
    seg_max_fix<<<cdiv(N_NODES * HEADS, 256), 256, 0, stream>>>(menc, mdec, N_NODES * HEADS);
    edge_exp<<<cdiv(N_EDGES * HEADS, 256), 256, 0, stream>>>(alpha, mdec, dst, denom);
    edge_aggregate<<<cdiv((long long)N_EDGES * HEADS * 32, 256), 256, 0, stream>>>(
        alpha, denom, v, src, dst, agg);
    combine_heads<<<cdiv(N_NODES * HID, 256), 256, 0, stream>>>(agg, skip, xout, N_NODES);
  };

  run_layer(xin,  F_INDIM, wq1, bq1, wk1, bk1, wv1, bv1, ws1, bs1, xcur);
  run_layer(xcur, HID,     wq2, bq2, wk2, bk2, wv2, bv2, ws2, bs2, xcur);

  fill_f32<<<2, 256, 0, stream>>>(bnsum, 0.f, 512);
  bn_reduce<<<80, 256, 0, stream>>>(xcur, bnsum, N_NODES);
  bn_finalize<<<1, 256, 0, stream>>>(bnsum, gam, bet, bninv, bnbet, (float)N_NODES);
  head_kernel<<<cdiv(N_NODES, 256), 256, 0, stream>>>(xcur, bninv, bnbet, lw, lb, out, N_NODES);
}
